// DecoderRNN_53137335386232
// MI455X (gfx1250) — matrix operand order repaired and checked by reading, not compiled, not run on hardware
//
#include <hip/hip_runtime.h>
#include <hip/hip_bf16.h>

typedef __attribute__((ext_vector_type(16))) _Float16 v16h;
typedef __attribute__((ext_vector_type(8)))  _Float16 v8h;
typedef __attribute__((ext_vector_type(8)))  float    v8f;

#define BATCH    65536
#define TILE     64
#define NTHREADS 256
#define HDIM     256
#define KDIM     512          // K = 256 (z) + 256 (h)
#define SSTEPS   4
#define NTA      4
#define NMA      11
#define LEN_ACT  16
#define ZSTRIDE  264          // 256 + 8 f16 pad; row = 528B (16B aligned, bank-skewed)

// d_out flat layout: [0,65536) log_p | [65536,589824) t_actions |
// [589824,1114112) m_actions | 1114112 t_ent | 1114113 m_ent
#define OUT_TACT   65536
#define OUT_MACT   (65536 + 524288)
#define OUT_ENT_T  1114112
#define OUT_ENT_M  1114113

union V16U { v16h v; v8h h[2]; };
__device__ __forceinline__ v16h mk16(v8h lo, v8h hi) { V16U u; u.h[0] = lo; u.h[1] = hi; return u.v; }
__device__ __forceinline__ float sigf(float x) { return 1.0f / (1.0f + __expf(-x)); }

// Index of the action embedding feeding this LSTM call (pure function of inputs).
__device__ __forceinline__ int prevAction(int grow, int branch, int s, int p,
                                          const int* __restrict__ tA,
                                          const int* __restrict__ mA) {
  if (p == 1) return tA[grow * 8 + branch * 4 + s];       // second call uses t-action
  if (s > 0)  return mA[grow * 8 + branch * 4 + (s - 1)]; // carry: prev = last m-action
  if (branch == 1) return mA[grow * 8 + 3];               // carried across branches
  return LEN_ACT - 1;                                     // initial prev = 15
}

// One K=32 slice of the 4-gate WMMA update. ap: lane-adjusted A row pointer.
// bbase = WB + nl*KDIM + kGlobal + hiSel*16 (gate g adds g*256*KDIM).
__device__ __forceinline__ void gateStep(const _Float16* __restrict__ ap,
                                         const _Float16* __restrict__ bbase,
                                         v8f& ai, v8f& af, v8f& ag, v8f& ao) {
  const v16h aF = mk16(*(const v8h*)ap, *(const v8h*)(ap + 8));
  const v16h b0 = mk16(*(const v8h*)(bbase),
                       *(const v8h*)(bbase + 8));
  const v16h b1 = mk16(*(const v8h*)(bbase + (size_t)256 * KDIM),
                       *(const v8h*)(bbase + (size_t)256 * KDIM + 8));
  const v16h b2 = mk16(*(const v8h*)(bbase + (size_t)512 * KDIM),
                       *(const v8h*)(bbase + (size_t)512 * KDIM + 8));
  const v16h b3 = mk16(*(const v8h*)(bbase + (size_t)768 * KDIM),
                       *(const v8h*)(bbase + (size_t)768 * KDIM + 8));
  ai = __builtin_amdgcn_wmma_f32_16x16x32_f16(false, aF, false, b0, (short)0, ai, false, false);
  af = __builtin_amdgcn_wmma_f32_16x16x32_f16(false, aF, false, b1, (short)0, af, false, false);
  ag = __builtin_amdgcn_wmma_f32_16x16x32_f16(false, aF, false, b2, (short)0, ag, false, false);
  ao = __builtin_amdgcn_wmma_f32_16x16x32_f16(false, aF, false, b3, (short)0, ao, false, false);
}

// ---------------------------------------------------------------------------
// Prep: build f16 weight matrix WB[n][k] = [W_ih_z | W_hh] (N-major, K inner)
// plus constT[4][1024] (bias + branch + actionid contributions) and
// actT[16][1024] (action-embedding contributions).
// ---------------------------------------------------------------------------
__global__ void decoder_prep(const float* __restrict__ action_emb,
                             const float* __restrict__ branch_emb,
                             const float* __restrict__ actionid_emb,
                             const float* __restrict__ W_ih,
                             const float* __restrict__ W_hh,
                             const float* __restrict__ b_ih,
                             const float* __restrict__ b_hh,
                             _Float16* __restrict__ WB,
                             float* __restrict__ constT,
                             float* __restrict__ actT) {
  const int n = blockIdx.x * blockDim.x + threadIdx.x;
  if (n >= 1024) return;
  for (int k = 0; k < 256; ++k) WB[(size_t)n * KDIM + k]       = (_Float16)W_ih[n * 448 + k];
  for (int k = 0; k < 256; ++k) WB[(size_t)n * KDIM + 256 + k] = (_Float16)W_hh[n * 256 + k];
  const float bias = b_ih[n] + b_hh[n];
  float sb[2], sp[2];
  for (int b2 = 0; b2 < 2; ++b2) {
    float s1 = 0.0f, s2 = 0.0f;
    for (int e = 0; e < 64; ++e) {
      s1 += branch_emb[b2 * 64 + e]   * W_ih[n * 448 + 256 + e];
      s2 += actionid_emb[b2 * 64 + e] * W_ih[n * 448 + 320 + e];
    }
    sb[b2] = s1; sp[b2] = s2;
  }
  for (int b2 = 0; b2 < 2; ++b2)
    for (int p2 = 0; p2 < 2; ++p2)
      constT[(b2 * 2 + p2) * 1024 + n] = bias + sb[b2] + sp[p2];
  for (int a = 0; a < LEN_ACT; ++a) {
    float sa = 0.0f;
    for (int e = 0; e < 64; ++e) sa += action_emb[a * 64 + e] * W_ih[n * 448 + 384 + e];
    actT[a * 1024 + n] = sa;
  }
}

// ---------------------------------------------------------------------------
// Copy int action outputs (as float) into d_out.
// ---------------------------------------------------------------------------
__global__ void decoder_init_out(const int* __restrict__ tA, const int* __restrict__ mA,
                                 float* __restrict__ out) {
  const int i = blockIdx.x * blockDim.x + threadIdx.x;
  if (i < 524288) {
    out[OUT_TACT + i] = (float)tA[i];
    out[OUT_MACT + i] = (float)mA[i];
  }
}

// ---------------------------------------------------------------------------
// Main persistent-tile kernel: 1024 blocks x 256 threads (8 wave32).
// Each block runs the whole 16-call LSTM sequence for 64 batch rows.
// ---------------------------------------------------------------------------
__global__ __launch_bounds__(NTHREADS)
void decoder_main(const float* __restrict__ z1, const float* __restrict__ z2,
                  const int* __restrict__ tA, const int* __restrict__ mA,
                  const float* __restrict__ W_t, const float* __restrict__ b_t,
                  const float* __restrict__ W_m, const float* __restrict__ b_m,
                  const _Float16* __restrict__ WB,
                  const float* __restrict__ constT, const float* __restrict__ actT,
                  float* __restrict__ out, float* __restrict__ entPartial) {
  __shared__ __align__(16) _Float16 sZ[TILE][ZSTRIDE];
  __shared__ __align__(16) _Float16 sH[2][TILE][ZSTRIDE];  // double-buffered hidden state
  __shared__ float entSh[2 * TILE];

  const int tid  = threadIdx.x;
  const int lane = tid & 31;
  const int wv   = tid >> 5;
  const int blockStart = blockIdx.x * TILE;
  const int hiSel = lane >> 4;                 // 0 for lanes 0-15, 1 for 16-31

  // LSTM cell state c, f32, held in registers (static wave/tile ownership).
  float cReg[8][8];
  #pragma unroll
  for (int j = 0; j < 8; ++j)
    #pragma unroll
    for (int r = 0; r < 8; ++r) cReg[j][r] = 0.0f;

  // zero initial hidden buffer
  for (int i = tid; i < TILE * ZSTRIDE; i += NTHREADS) (&sH[0][0][0])[i] = (_Float16)0.0f;

  int hin = 0, hout = 1;
  float lp = 0.0f, entT = 0.0f, entM = 0.0f;
  int maskReg = 0xF;

  for (int branch = 0; branch < 2; ++branch) {
    const float* __restrict__ z = branch ? z2 : z1;
    __syncthreads();
    for (int i = tid; i < TILE * HDIM; i += NTHREADS) {
      const int r = i >> 8, k = i & 255;
      sZ[r][k] = (_Float16)z[(size_t)(blockStart + r) * HDIM + k];
    }
    maskReg = 0xF;
    __syncthreads();

    for (int s = 0; s < SSTEPS; ++s) {
      for (int p = 0; p < 2; ++p) {
        const int cIdx = branch * 2 + p;

        // ---------------- matmul phase: gates = [z,h] @ WBᵀ ----------------
        #pragma unroll
        for (int j = 0; j < 8; ++j) {
          const int Ct = 2 * wv + (j >> 2);      // H column tile (16 cols)
          const int Mt = j & 3;                  // batch-row tile (16 rows)
          const int nl = Ct * 16 + (lane & 15);  // this lane's H column
          const int arow = Mt * 16 + (lane & 15);

          v8f ai = {0,0,0,0,0,0,0,0};
          v8f af = ai, ag = ai, ao = ai;

          const _Float16* aZ = &sZ[arow][hiSel ? 16 : 0];         // z half, K 0..255
          const _Float16* aH = &sH[hin][arow][hiSel ? 16 : 0];    // h half, K 256..511
          const _Float16* bN = WB + (size_t)nl * KDIM + (hiSel << 4);

          #pragma unroll 2
          for (int k0 = 0; k0 < 256; k0 += 32)            // z contribution
            gateStep(aZ + k0, bN + k0, ai, af, ag, ao);
          #pragma unroll 2
          for (int k0 = 0; k0 < 256; k0 += 32)            // h contribution
            gateStep(aH + k0, bN + 256 + k0, ai, af, ag, ao);

          // ------- epilogue: + const/action tables, LSTM cell update -------
          const float ci0 = constT[cIdx * 1024 +       nl];
          const float cf0 = constT[cIdx * 1024 + 256 + nl];
          const float cg0 = constT[cIdx * 1024 + 512 + nl];
          const float co0 = constT[cIdx * 1024 + 768 + nl];
          #pragma unroll
          for (int r = 0; r < 8; ++r) {
            const int rm = Mt * 16 + r + (hiSel << 3);   // C/D layout: M = r (+8 hi half)
            const int grow = blockStart + rm;
            const int pv = prevAction(grow, branch, s, p, tA, mA);
            const float gi = ai[r] + ci0 + actT[pv * 1024 +       nl];
            const float gf = af[r] + cf0 + actT[pv * 1024 + 256 + nl];
            const float gg = ag[r] + cg0 + actT[pv * 1024 + 512 + nl];
            const float go = ao[r] + co0 + actT[pv * 1024 + 768 + nl];
            const float cN = sigf(gf) * cReg[j][r] + sigf(gi) * tanhf(gg);
            const float hN = sigf(go) * tanhf(cN);
            cReg[j][r] = cN;
            sH[hout][rm][nl] = (_Float16)hN;
          }
        }
        __syncthreads();

        // ---------- per-row heads: masked log-softmax + entropy ----------
        if (tid < TILE) {
          const int grow = blockStart + tid;
          const _Float16* hrow = &sH[hout][tid][0];
          if (p == 0) {
            const int ta = tA[grow * 8 + branch * 4 + s];
            float l[NTA] = { b_t[0], b_t[1], b_t[2], b_t[3] };
            for (int k = 0; k < HDIM; ++k) {
              const float hk = (float)hrow[k];
              #pragma unroll
              for (int q2 = 0; q2 < NTA; ++q2) l[q2] += hk * W_t[q2 * HDIM + k];
            }
            #pragma unroll
            for (int q2 = 0; q2 < NTA; ++q2)
              if (!(maskReg & (1 << q2))) l[q2] = -1.0e9f;
            const float mx = fmaxf(fmaxf(l[0], l[1]), fmaxf(l[2], l[3]));
            float se = 0.0f;
            #pragma unroll
            for (int q2 = 0; q2 < NTA; ++q2) se += __expf(l[q2] - mx);
            const float lse = mx + __logf(se);
            lp += l[ta] - lse;
            float e = 0.0f;
            #pragma unroll
            for (int q2 = 0; q2 < NTA; ++q2) {
              const float d = l[q2] - lse;
              const float pj = __expf(d);
              if (pj > 0.0f) e -= pj * d;
            }
            entT += e;
            maskReg &= ~(1 << ta);
          } else {
            const int ma = mA[grow * 8 + branch * 4 + s];
            float l[NMA];
            #pragma unroll
            for (int q2 = 0; q2 < NMA; ++q2) l[q2] = b_m[q2];
            for (int k = 0; k < HDIM; ++k) {
              const float hk = (float)hrow[k];
              #pragma unroll
              for (int q2 = 0; q2 < NMA; ++q2) l[q2] += hk * W_m[q2 * HDIM + k];
            }
            float mx = l[0];
            #pragma unroll
            for (int q2 = 1; q2 < NMA; ++q2) mx = fmaxf(mx, l[q2]);
            float se = 0.0f;
            #pragma unroll
            for (int q2 = 0; q2 < NMA; ++q2) se += __expf(l[q2] - mx);
            const float lse = mx + __logf(se);
            lp += l[ma] - lse;
            float e = 0.0f;
            #pragma unroll
            for (int q2 = 0; q2 < NMA; ++q2) {
              const float d = l[q2] - lse;
              e -= __expf(d) * d;
            }
            entM += e;
          }
        }
        hin ^= 1; hout ^= 1;
        __syncthreads();
      }
    }
  }

  if (tid < TILE) {
    out[blockStart + tid] = lp;     // log_p
    entSh[tid] = entT;
    entSh[TILE + tid] = entM;
  }
  __syncthreads();
  if (tid == 0) {
    float st = 0.0f, sm = 0.0f;
    for (int i = 0; i < TILE; ++i) { st += entSh[i]; sm += entSh[TILE + i]; }
    entPartial[blockIdx.x]        = st;   // deterministic per-block partials
    entPartial[1024 + blockIdx.x] = sm;
  }
}

// ---------------------------------------------------------------------------
// Deterministic entropy reduction (avoids float atomics for replay stability).
// ---------------------------------------------------------------------------
__global__ void decoder_finalize(const float* __restrict__ entPartial, float* __restrict__ out) {
  if (threadIdx.x == 0 && blockIdx.x == 0) {
    float st = 0.0f, sm = 0.0f;
    for (int i = 0; i < 1024; ++i) { st += entPartial[i]; sm += entPartial[1024 + i]; }
    const float scale = 1.0f / (8.0f * (float)BATCH);   // mean over batch, / (2*S)
    out[OUT_ENT_T] = st * scale;
    out[OUT_ENT_M] = sm * scale;
  }
}

extern "C" void kernel_launch(void* const* d_in, const int* in_sizes, int n_in,
                              void* d_out, int out_size, void* d_ws, size_t ws_size,
                              hipStream_t stream) {
  const float* z1           = (const float*)d_in[0];
  const float* z2           = (const float*)d_in[1];
  const int*   tA           = (const int*)d_in[2];
  const int*   mA           = (const int*)d_in[3];
  const float* action_emb   = (const float*)d_in[4];
  const float* branch_emb   = (const float*)d_in[5];
  const float* actionid_emb = (const float*)d_in[6];
  const float* W_ih         = (const float*)d_in[7];
  const float* W_hh         = (const float*)d_in[8];
  const float* b_ih         = (const float*)d_in[9];
  const float* b_hh         = (const float*)d_in[10];
  const float* W_t          = (const float*)d_in[11];
  const float* b_t          = (const float*)d_in[12];
  const float* W_m          = (const float*)d_in[13];
  const float* b_m          = (const float*)d_in[14];
  float* out = (float*)d_out;

  // workspace layout: WB f16[1024][512] | constT f32[4][1024] | actT f32[16][1024] | entPartial f32[2048]
  _Float16* WB     = (_Float16*)d_ws;
  float* constT    = (float*)((char*)d_ws + (size_t)1024 * KDIM * sizeof(_Float16));
  float* actT      = constT + 4 * 1024;
  float* entPart   = actT + LEN_ACT * 1024;

  decoder_init_out<<<524288 / NTHREADS, NTHREADS, 0, stream>>>(tA, mA, out);
  decoder_prep<<<4, NTHREADS, 0, stream>>>(action_emb, branch_emb, actionid_emb,
                                           W_ih, W_hh, b_ih, b_hh, WB, constT, actT);
  decoder_main<<<BATCH / TILE, NTHREADS, 0, stream>>>(z1, z2, tA, mA, W_t, b_t, W_m, b_m,
                                                      WB, constT, actT, out, entPart);
  decoder_finalize<<<1, 1, 0, stream>>>(entPart, out);
}
